// WindowedAttention_24068996727078
// MI455X (gfx1250) — matrix operand order repaired and checked by reading, not compiled, not run on hardware
//
#include <hip/hip_runtime.h>

// Problem constants (from reference)
#define WSZ 128   // window width
#define NH  8     // heads
#define MD  1024  // model dim D
#define DKC 512   // dim key/value DK
#define DH  64    // per-head dim
#define BBATCH 8
#define SS  4096
#define NP  32    // windows per batch (SS/WSZ)
#define BN  256   // total windows (BBATCH*NP)

typedef __bf16 bf16_t;
typedef __attribute__((ext_vector_type(16))) __bf16 v16bf;
typedef __attribute__((ext_vector_type(8)))  __bf16 v8bf;
typedef __attribute__((ext_vector_type(8)))  float  v8f;

static __device__ __forceinline__ bf16_t f2bf(float f) {
  unsigned u = __builtin_bit_cast(unsigned, f);
  u += 0x7FFFu + ((u >> 16) & 1u);           // round-to-nearest-even
  unsigned short h = (unsigned short)(u >> 16);
  return __builtin_bit_cast(bf16_t, h);
}

static __device__ __forceinline__ v16bf frag16(v8bf lo, v8bf hi) {
  return __builtin_shufflevector(lo, hi, 0,1,2,3,4,5,6,7,8,9,10,11,12,13,14,15);
}

// A fragment: 16x32 bf16, row-major source; `row` already points at this
// lane's M-row base. K elements: {hl*8..+7} and {16+hl*8..+7}.
static __device__ __forceinline__ v16bf ldfrag_a(const bf16_t* row, int kk, int hl) {
  v8bf lo = *(const v8bf*)(row + kk + hl * 8);
  v8bf hi = *(const v8bf*)(row + kk + 16 + hl * 8);
  return frag16(lo, hi);
}

// B fragment: 32x16 bf16 from [N][K] storage; `colrow` points at this lane's
// N-column row base. K elements: {hl*8..+7} and {16+hl*8..+7}.
static __device__ __forceinline__ v16bf ldfrag_b(const bf16_t* colrow, int kk, int hl) {
  const bf16_t* p = colrow + kk + hl * 8;
  return frag16(*(const v8bf*)p, *(const v8bf*)(p + 16));
}

static __device__ __forceinline__ v8f wmma_bf(v16bf a, v16bf b, v8f c) {
  return __builtin_amdgcn_wmma_f32_16x16x32_bf16(false, a, false, b, (short)0, c,
                                                 false, false);
}

// ---------------- K0: f32 -> bf16 weight conversion ----------------
__global__ void __launch_bounds__(256) k_cvt(const float* __restrict__ src,
                                             bf16_t* __restrict__ dst, int n) {
  int i = blockIdx.x * blockDim.x + threadIdx.x;
  if (i < n) dst[i] = f2bf(src[i]);
}

// ---------------- K1: QKV projection ----------------
// grid (BN, DKC/128), block 256 (8 waves); each wave: 16 channel rows x 128 cols.
// Outputs: qT,kT as [BN][H][W][DH] (position-major), vN as [BN][H][DH][W].
__global__ void __launch_bounds__(256) k_qkv(
    const float* __restrict__ qk, const float* __restrict__ vin,
    const bf16_t* __restrict__ Wqkv,
    const float* __restrict__ bq, const float* __restrict__ bk,
    const float* __restrict__ bv,
    bf16_t* __restrict__ qT, bf16_t* __restrict__ kT, bf16_t* __restrict__ vN) {
  __shared__ bf16_t lx[WSZ * 32];          // X^T tile: [w][k] bf16
  const int bw = blockIdx.x;
  const int e0 = blockIdx.y * 128;
  const int b = bw >> 5, p = bw & 31, s0 = p * WSZ;
  const int t = threadIdx.x, lane = t & 31, wv = t >> 5;
  const int hl = lane >> 4, ln = lane & 15;
  const int erow = e0 + wv * 16 + ln;

  const float*  Xg = qk + (size_t)b * MD * SS + s0;
  const bf16_t* Aq = Wqkv + (size_t)erow * MD;
  const bf16_t* Ak = Wqkv + ((size_t)DKC + erow) * MD;

  v8f accQ[8], accK[8];
  for (int j = 0; j < 8; ++j) { accQ[j] = {}; accK[j] = {}; }

  const int sw = t & 127, sd = (t >> 7) * 16;
  for (int kk = 0; kk < MD; kk += 32) {
    for (int i = 0; i < 16; ++i)
      lx[sw * 32 + sd + i] = f2bf(Xg[(size_t)(kk + sd + i) * SS + sw]);
    __syncthreads();
    v16bf aq = ldfrag_a(Aq, kk, hl);
    v16bf ak = ldfrag_a(Ak, kk, hl);
    for (int j = 0; j < 8; ++j) {
      v16bf bx = ldfrag_b(&lx[(j * 16 + ln) * 32], 0, hl);
      accQ[j] = wmma_bf(aq, bx, accQ[j]);
      accK[j] = wmma_bf(ak, bx, accK[j]);
    }
    __syncthreads();
  }
  for (int j = 0; j < 8; ++j) {
    int w = j * 16 + ln;
    for (int r = 0; r < 8; ++r) {
      int e = e0 + wv * 16 + r + 8 * hl;
      int head = e >> 6, eh = e & 63;
      size_t o = (((size_t)bw * NH + head) * WSZ + w) * DH + eh;
      qT[o] = f2bf((accQ[j][r] + bq[e]) * 0.125f);  // fold 1/sqrt(dh)
      kT[o] = f2bf(accK[j][r] + bk[e]);
    }
  }

  // ---- V ----
  for (int j = 0; j < 8; ++j) accQ[j] = {};
  const float*  Vg = vin + (size_t)b * MD * SS + s0;
  const bf16_t* Av = Wqkv + ((size_t)2 * DKC + erow) * MD;
  __syncthreads();
  for (int kk = 0; kk < MD; kk += 32) {
    for (int i = 0; i < 16; ++i)
      lx[sw * 32 + sd + i] = f2bf(Vg[(size_t)(kk + sd + i) * SS + sw]);
    __syncthreads();
    v16bf av = ldfrag_a(Av, kk, hl);
    for (int j = 0; j < 8; ++j) {
      v16bf bx = ldfrag_b(&lx[(j * 16 + ln) * 32], 0, hl);
      accQ[j] = wmma_bf(av, bx, accQ[j]);
    }
    __syncthreads();
  }
  for (int j = 0; j < 8; ++j) {
    int w = j * 16 + ln;
    for (int r = 0; r < 8; ++r) {
      int e = e0 + wv * 16 + r + 8 * hl;
      int head = e >> 6, eh = e & 63;
      vN[(((size_t)bw * NH + head) * DH + eh) * WSZ + w] = f2bf(accQ[j][r] + bv[e]);
    }
  }
}

// ---------------- K2: attention core (per window, per head) ----------------
// grid (BN, NH), block 256; dynamic LDS = 128*132 floats for scores.
__global__ void __launch_bounds__(256) k_attn(
    const bf16_t* __restrict__ qT, const bf16_t* __restrict__ kT,
    const bf16_t* __restrict__ vN, const float* __restrict__ masks,
    bf16_t* __restrict__ oT) {
  extern __shared__ float sc[];            // [128][132] padded scores
  const int bw = blockIdx.x, h = blockIdx.y;
  const int b = bw >> 5, p = bw & 31, s0 = p * WSZ;
  const int t = threadIdx.x, lane = t & 31, wv = t >> 5;
  const int hl = lane >> 4, ln = lane & 15;
  const int i0 = wv * 16;

  const bf16_t* qb = qT + ((size_t)bw * NH + h) * WSZ * DH;
  const bf16_t* kb = kT + ((size_t)bw * NH + h) * WSZ * DH;
  const bf16_t* vb = vN + ((size_t)bw * NH + h) * DH * WSZ;

  // scores = (q/sqrt(dh))^T k  : M=i, N=j, K=d (64)
  v8f acc[8];
  for (int j = 0; j < 8; ++j) acc[j] = {};
  for (int kk = 0; kk < DH; kk += 32) {
    v16bf a = ldfrag_a(qb + (size_t)(i0 + ln) * DH, kk, hl);
    for (int j = 0; j < 8; ++j) {
      v16bf bb = ldfrag_b(kb + (size_t)(j * 16 + ln) * DH, kk, hl);
      acc[j] = wmma_bf(a, bb, acc[j]);
    }
  }
  for (int j = 0; j < 8; ++j) {
    int jj = j * 16 + ln;
    for (int r = 0; r < 8; ++r)
      sc[(size_t)(i0 + r + 8 * hl) * 132 + jj] = acc[j][r];
  }
  __syncthreads();

  // masked softmax over keys, one thread per query row, in f32; re-pack bf16
  // in place at the head of each padded row.
  if (t < WSZ) {
    float* row = sc + (size_t)t * 132;
    const float* mrow = masks + (size_t)b * SS + s0;
    float mx = -3.0e38f;
    for (int j = 0; j < WSZ; ++j) {
      float x = (mrow[j] > 0.f) ? row[j] : -1.0e9f;
      row[j] = x;
      mx = fmaxf(mx, x);
    }
    float sum = 0.f;
    for (int j = 0; j < WSZ; ++j) { float e = expf(row[j] - mx); row[j] = e; sum += e; }
    float inv = 1.f / sum;
    bf16_t* arow = (bf16_t*)row;           // in-place: bf16[j] < f32[j] byte span
    for (int j = 0; j < WSZ; ++j) arow[j] = f2bf(row[j] * inv);
  }
  __syncthreads();

  // o = attn @ v^T : M=i, N=d (64), K=j (128); then exact GELU, store [W][DK]
  v8f oa[4];
  for (int d = 0; d < 4; ++d) oa[d] = {};
  for (int kk = 0; kk < WSZ; kk += 32) {
    v16bf a = ldfrag_a((const bf16_t*)(sc + (size_t)(i0 + ln) * 132), kk, hl);
    for (int dt = 0; dt < 4; ++dt) {
      v16bf bb = ldfrag_b(vb + (size_t)(dt * 16 + ln) * WSZ, kk, hl);
      oa[dt] = wmma_bf(a, bb, oa[dt]);
    }
  }
  for (int dt = 0; dt < 4; ++dt) {
    int d = dt * 16 + ln;
    for (int r = 0; r < 8; ++r) {
      int i = i0 + r + 8 * hl;
      float x = oa[dt][r];
      float g = 0.5f * x * (1.f + erff(x * 0.70710678118f));
      oT[((size_t)bw * WSZ + i) * DKC + h * DH + d] = f2bf(g);
    }
  }
}

// ---------------- K3: output projection (global-fed WMMA) ----------------
// grid (BN, MD/128), block 256. out[de,w] = Wo @ gelu_o + bo, * mask
__global__ void __launch_bounds__(256) k_out(
    const bf16_t* __restrict__ WoB, const bf16_t* __restrict__ oT,
    const float* __restrict__ bo, const float* __restrict__ masks,
    float* __restrict__ out) {
  const int bw = blockIdx.x;
  const int de0 = blockIdx.y * 128;
  const int b = bw >> 5, p = bw & 31, s0 = p * WSZ;
  const int t = threadIdx.x, lane = t & 31, wv = t >> 5;
  const int hl = lane >> 4, ln = lane & 15;

  const bf16_t* Ar = WoB + (size_t)(de0 + wv * 16 + ln) * DKC;
  const bf16_t* Bb = oT + (size_t)bw * WSZ * DKC;

  v8f acc[8];
  for (int j = 0; j < 8; ++j) acc[j] = {};
  for (int kk = 0; kk < DKC; kk += 32) {
    v16bf a = ldfrag_a(Ar, kk, hl);
    for (int j = 0; j < 8; ++j) {
      v16bf bb = ldfrag_b(Bb + (size_t)(j * 16 + ln) * DKC, kk, hl);
      acc[j] = wmma_bf(a, bb, acc[j]);
    }
  }
  for (int j = 0; j < 8; ++j) {
    int w = j * 16 + ln;
    float m = masks[(size_t)b * SS + s0 + w];
    for (int r = 0; r < 8; ++r) {
      int de = de0 + wv * 16 + r + 8 * hl;
      out[((size_t)b * MD + de) * SS + s0 + w] = (acc[j][r] + bo[de]) * m;
    }
  }
}

extern "C" void kernel_launch(void* const* d_in, const int* in_sizes, int n_in,
                              void* d_out, int out_size, void* d_ws, size_t ws_size,
                              hipStream_t stream) {
  (void)in_sizes; (void)n_in; (void)out_size; (void)ws_size;
  const float* qk    = (const float*)d_in[0];
  const float* vin   = (const float*)d_in[1];
  const float* masks = (const float*)d_in[2];
  const float* Wq = (const float*)d_in[3];
  const float* bq = (const float*)d_in[4];
  const float* Wk = (const float*)d_in[5];
  const float* bk = (const float*)d_in[6];
  const float* Wv = (const float*)d_in[7];
  const float* bv = (const float*)d_in[8];
  const float* Wo = (const float*)d_in[9];
  const float* bo = (const float*)d_in[10];
  float* out = (float*)d_out;

  char* ws = (char*)d_ws;
  bf16_t* Wqkv_bf = (bf16_t*)ws;  ws += (size_t)3 * DKC * MD * sizeof(bf16_t);
  bf16_t* Wo_bf   = (bf16_t*)ws;  ws += (size_t)MD * DKC * sizeof(bf16_t);
  bf16_t* qT      = (bf16_t*)ws;  ws += (size_t)BN * DKC * WSZ * sizeof(bf16_t);
  bf16_t* kT      = (bf16_t*)ws;  ws += (size_t)BN * DKC * WSZ * sizeof(bf16_t);
  bf16_t* vN      = (bf16_t*)ws;  ws += (size_t)BN * DKC * WSZ * sizeof(bf16_t);
  bf16_t* oT      = (bf16_t*)ws;

  const int nW = DKC * MD;
  k_cvt<<<(nW + 255) / 256, 256, 0, stream>>>(Wq, Wqkv_bf, nW);
  k_cvt<<<(nW + 255) / 256, 256, 0, stream>>>(Wk, Wqkv_bf + nW, nW);
  k_cvt<<<(nW + 255) / 256, 256, 0, stream>>>(Wv, Wqkv_bf + 2 * nW, nW);
  k_cvt<<<(nW + 255) / 256, 256, 0, stream>>>(Wo, Wo_bf, nW);

  k_qkv<<<dim3(BN, DKC / 128), 256, 0, stream>>>(qk, vin, Wqkv_bf, bq, bk, bv,
                                                 qT, kT, vN);
  k_attn<<<dim3(BN, NH), 256, (size_t)128 * 132 * sizeof(float), stream>>>(
      qT, kT, vN, masks, oT);
  k_out<<<dim3(BN, MD / 128), 256, 0, stream>>>(Wo_bf, oT, bo, masks, out);
}
